// YatNMN_19791209300370
// MI455X (gfx1250) — matrix operand order repaired and checked by reading, not compiled, not run on hardware
//
#include <hip/hip_runtime.h>
#include <hip/hip_bf16.h>
#include <math.h>

// ---------------------------------------------------------------------------
// YatNMN fused layer for gfx1250 (MI455X), wave32 + WMMA + async LDS copies.
//
//   y0 = X @ W          (M=8192, K=2048, N=2048, f32 in/out)
//   d  = xsq[m] + wsq[n] - 2*y0
//   y  = (y0*y0)/(d+1e-5) + bias[n]
//   out= y * (sqrt(2048)/log(2049))^alpha
//
// Pipeline:
//   1) prepass: split X,W into bf16 hi/lo pairs (Xh/Xl row-major [M][K],
//      Wh/Wl transposed [N][K]) + row/col squared norms -> workspace.
//   2) GEMM: double-buffered global_load_async_to_lds_b128 staging,
//      ds_load_b128 fragment fetches, 12x v_wmma_f32_16x16x32_bf16 per
//      K-step (hi*hi + lo*hi + hi*lo f32 accumulation), fused Yat epilogue.
// ---------------------------------------------------------------------------

typedef __attribute__((ext_vector_type(16))) __bf16 v16bf;
typedef __attribute__((ext_vector_type(4)))  __bf16 v4bf;
typedef __attribute__((ext_vector_type(8)))  float  v8f;

#define M_TOTAL 8192
#define N_TOTAL 2048
#define K_TOTAL 2048
#define BM 128
#define BN 64
#define BK 32

// LDS layout (bytes), per buffer: padded row stride 80B for 32 bf16 of K.
#define ROWB 80
#define AH_OFF 0
#define AL_OFF (BM * ROWB)              // 10240
#define BH_OFF (2 * BM * ROWB)          // 20480
#define BL_OFF (2 * BM * ROWB + BN * ROWB)  // 25600
#define BUF_BYTES (2 * BM * ROWB + 2 * BN * ROWB)  // 30720

union FragU { v16bf v; uint4 q[2]; };

// ---- CDNA5 async global->LDS copy (ASYNCcnt-tracked), GVS addressing ------
__device__ __forceinline__ void async_ld_b128(unsigned lds_byte_off,
                                              unsigned gl_byte_off,
                                              const void* sbase) {
  asm volatile("global_load_async_to_lds_b128 %0, %1, %2"
               :
               : "v"(lds_byte_off), "v"(gl_byte_off),
                 "s"((unsigned long long)(uintptr_t)sbase)
               : "memory");
}

#define WAIT_ASYNCCNT(N) asm volatile("s_wait_asynccnt %0" ::"n"(N) : "memory")

// ---- prepass: X -> Xh/Xl bf16 [M][K] + xsq[m] -----------------------------
__global__ __launch_bounds__(256) void yat_convx_kernel(
    const float* __restrict__ X, __bf16* __restrict__ Xh,
    __bf16* __restrict__ Xl, float* __restrict__ xsq) {
  __shared__ float red[256];
  const int row = blockIdx.x;
  const float4* src = reinterpret_cast<const float4*>(X + (size_t)row * K_TOTAL);
  float s = 0.0f;
#pragma unroll
  for (int i = 0; i < 2; ++i) {
    const int idx = threadIdx.x + i * 256;  // float4 index within row (512 total)
    const float4 v = src[idx];
    s += v.x * v.x + v.y * v.y + v.z * v.z + v.w * v.w;
    const __bf16 h0 = (__bf16)v.x, h1 = (__bf16)v.y, h2 = (__bf16)v.z, h3 = (__bf16)v.w;
    const v4bf vh = {h0, h1, h2, h3};
    const v4bf vl = {(__bf16)(v.x - (float)h0), (__bf16)(v.y - (float)h1),
                     (__bf16)(v.z - (float)h2), (__bf16)(v.w - (float)h3)};
    const size_t off = (size_t)row * K_TOTAL + (size_t)idx * 4;
    *reinterpret_cast<v4bf*>(Xh + off) = vh;
    *reinterpret_cast<v4bf*>(Xl + off) = vl;
  }
  red[threadIdx.x] = s;
  __syncthreads();
  for (int off = 128; off > 0; off >>= 1) {
    if (threadIdx.x < off) red[threadIdx.x] += red[threadIdx.x + off];
    __syncthreads();
  }
  if (threadIdx.x == 0) xsq[row] = red[0];
}

// ---- prepass: W[k][n] -> Wh/Wl bf16 transposed [N][K] (32x32 LDS tiles) ---
__global__ __launch_bounds__(256) void yat_convw_kernel(
    const float* __restrict__ W, __bf16* __restrict__ Wh,
    __bf16* __restrict__ Wl) {
  __shared__ __bf16 th[32][34];
  __shared__ __bf16 tl[32][34];
  const int k0 = blockIdx.x * 32;
  const int n0 = blockIdx.y * 32;
#pragma unroll
  for (int i = 0; i < 4; ++i) {
    const int idx = threadIdx.x + i * 256;
    const int r = idx >> 5, c = idx & 31;  // r: k-local, c: n-local
    const float v = W[(size_t)(k0 + r) * N_TOTAL + n0 + c];
    const __bf16 h = (__bf16)v;
    th[r][c] = h;
    tl[r][c] = (__bf16)(v - (float)h);
  }
  __syncthreads();
#pragma unroll
  for (int i = 0; i < 4; ++i) {
    const int idx = threadIdx.x + i * 256;
    const int r = idx >> 5, c = idx & 31;  // r: n-local, c: k-local
    const size_t off = (size_t)(n0 + r) * K_TOTAL + k0 + c;
    Wh[off] = th[c][r];
    Wl[off] = tl[c][r];
  }
}

// ---- prepass: wsq[n] = sum_k W[k][n]^2 ------------------------------------
__global__ __launch_bounds__(256) void yat_wsq_kernel(const float* __restrict__ W,
                                                      float* __restrict__ wsq) {
  const int j = blockIdx.x * 256 + threadIdx.x;
  float s = 0.0f;
  for (int k = 0; k < K_TOTAL; ++k) {
    const float v = W[(size_t)k * N_TOTAL + j];
    s += v * v;
  }
  wsq[j] = s;
}

// ---- fused GEMM + Yat epilogue --------------------------------------------
__global__ __launch_bounds__(256) void yat_gemm_kernel(
    const __bf16* __restrict__ Xh, const __bf16* __restrict__ Xl,
    const __bf16* __restrict__ Wh, const __bf16* __restrict__ Wl,
    const float* __restrict__ bias, const float* __restrict__ alpha,
    const float* __restrict__ xsq, const float* __restrict__ wsq,
    float* __restrict__ out) {
  __shared__ __align__(16) unsigned char lds[2][BUF_BYTES];

  const int tid  = threadIdx.x;
  const int lane = tid & 31;
  const int wv   = tid >> 5;           // 8 waves: 4 (M) x 2 (N)
  const int wave_m = (wv >> 1) * 32;
  const int wave_n = (wv & 1) * 32;
  const int m0 = blockIdx.y * BM;
  const int n0 = blockIdx.x * BN;

  const int hl  = lane >> 4;           // lane half (0/1)
  const int lm  = lane & 15;
  const int khA = hl * 8;              // A-fragment K offset (ISA 16-bit A layout)
  const int khB = hl * 8;              // B-fragment K offset (ISA 16-bit B layout)

  // LDS base byte offset of this block's buffers (generic ptr low 32 bits = LDS addr)
  const unsigned lds_base = (unsigned)(uintptr_t)(&lds[0][0]);

  // Per-thread async copy chunk coordinates.
  const int arow0 = tid >> 2, ac40 = tid & 3;              // A chunks 0..255
  const int arow1 = (tid + 256) >> 2, ac41 = tid & 3;      // A chunks 256..511
  const int brow  = tid >> 2, bc4 = tid & 3;               // B chunks 0..255

  auto issue_tile = [&](int kt, int buf) {
    const unsigned lb = lds_base + (unsigned)buf * BUF_BYTES;
    // A hi/lo: 512 chunks each, 2 per thread per matrix
    unsigned g0 = (unsigned)(((m0 + arow0) * K_TOTAL + kt + ac40 * 8) * 2);
    unsigned g1 = (unsigned)(((m0 + arow1) * K_TOTAL + kt + ac41 * 8) * 2);
    unsigned l0 = (unsigned)(arow0 * ROWB + ac40 * 16);
    unsigned l1 = (unsigned)(arow1 * ROWB + ac41 * 16);
    async_ld_b128(lb + AH_OFF + l0, g0, Xh);
    async_ld_b128(lb + AH_OFF + l1, g1, Xh);
    async_ld_b128(lb + AL_OFF + l0, g0, Xl);
    async_ld_b128(lb + AL_OFF + l1, g1, Xl);
    // B hi/lo: 256 chunks each, 1 per thread per matrix
    unsigned gb = (unsigned)(((n0 + brow) * K_TOTAL + kt + bc4 * 8) * 2);
    unsigned lbo = (unsigned)(brow * ROWB + bc4 * 16);
    async_ld_b128(lb + BH_OFF + lbo, gb, Wh);
    async_ld_b128(lb + BL_OFF + lbo, gb, Wl);
  };

  v8f acc[2][2] = {};

  auto compute_tile = [&](int buf) {
    const unsigned char* b = &lds[buf][0];
    FragU ah[2], al[2], bh[2], bl[2];
#pragma unroll
    for (int ms = 0; ms < 2; ++ms) {
      const int row = wave_m + ms * 16 + lm;
      const unsigned base = row * ROWB + khA * 2;
      ah[ms].q[0] = *reinterpret_cast<const uint4*>(b + AH_OFF + base);
      ah[ms].q[1] = *reinterpret_cast<const uint4*>(b + AH_OFF + base + 32);
      al[ms].q[0] = *reinterpret_cast<const uint4*>(b + AL_OFF + base);
      al[ms].q[1] = *reinterpret_cast<const uint4*>(b + AL_OFF + base + 32);
    }
#pragma unroll
    for (int ns = 0; ns < 2; ++ns) {
      const int col = wave_n + ns * 16 + lm;
      const unsigned base = col * ROWB + khB * 2;
      bh[ns].q[0] = *reinterpret_cast<const uint4*>(b + BH_OFF + base);
      bh[ns].q[1] = *reinterpret_cast<const uint4*>(b + BH_OFF + base + 32);
      bl[ns].q[0] = *reinterpret_cast<const uint4*>(b + BL_OFF + base);
      bl[ns].q[1] = *reinterpret_cast<const uint4*>(b + BL_OFF + base + 32);
    }
#pragma unroll
    for (int ms = 0; ms < 2; ++ms) {
#pragma unroll
      for (int ns = 0; ns < 2; ++ns) {
        acc[ms][ns] = __builtin_amdgcn_wmma_f32_16x16x32_bf16(
            false, ah[ms].v, false, bh[ns].v, (short)0, acc[ms][ns], false, false);
        acc[ms][ns] = __builtin_amdgcn_wmma_f32_16x16x32_bf16(
            false, al[ms].v, false, bh[ns].v, (short)0, acc[ms][ns], false, false);
        acc[ms][ns] = __builtin_amdgcn_wmma_f32_16x16x32_bf16(
            false, ah[ms].v, false, bl[ns].v, (short)0, acc[ms][ns], false, false);
      }
    }
  };

  // ---- double-buffered main loop ----
  issue_tile(0, 0);
  int cur = 0;
  for (int kt = 0; kt < K_TOTAL - BK; kt += BK) {
    issue_tile(kt + BK, cur ^ 1);   // prefetch next tile (overlaps with compute)
    WAIT_ASYNCCNT(6);               // 6 newest = next tile; current tile landed
    __syncthreads();                // all waves' current-tile data visible
    compute_tile(cur);
    __syncthreads();                // everyone done reading before next overwrite
    cur ^= 1;
  }
  WAIT_ASYNCCNT(0);
  __syncthreads();
  compute_tile(cur);

  // ---- fused Yat epilogue ----
  const float basev = sqrtf(2048.0f) / logf(2049.0f);  // constant-folded
  const float scale = __expf(alpha[0] * __logf(basev));

#pragma unroll
  for (int ms = 0; ms < 2; ++ms) {
#pragma unroll
    for (int ns = 0; ns < 2; ++ns) {
      const int n = n0 + wave_n + ns * 16 + lm;
      const float wn = wsq[n];
      const float bn = bias[n];
#pragma unroll
      for (int v = 0; v < 8; ++v) {
        const int m = m0 + wave_m + ms * 16 + hl * 8 + v;  // C/D layout: row = v + 8*half
        const float y0 = acc[ms][ns][v];
        const float dist = xsq[m] + wn - 2.0f * y0;
        const float y = (y0 * y0) / (dist + 1e-5f) + bn;
        out[(size_t)m * N_TOTAL + n] = y * scale;
      }
    }
  }
}

// ---------------------------------------------------------------------------
extern "C" void kernel_launch(void* const* d_in, const int* in_sizes, int n_in,
                              void* d_out, int out_size, void* d_ws, size_t ws_size,
                              hipStream_t stream) {
  const float* x     = (const float*)d_in[0];  // [4,2048,2048]
  const float* w     = (const float*)d_in[1];  // [2048,2048]
  const float* bias  = (const float*)d_in[2];  // [2048]
  const float* alpha = (const float*)d_in[3];  // [1]
  float* out = (float*)d_out;

  // workspace layout (~80 MB): xsq | wsq | Xh | Xl | Wh | Wl
  float* xsq = (float*)d_ws;                 // 8192 f32
  float* wsq = xsq + M_TOTAL;                // 2048 f32
  __bf16* Xh = (__bf16*)(wsq + N_TOTAL);
  __bf16* Xl = Xh + (size_t)M_TOTAL * K_TOTAL;
  __bf16* Wh = Xl + (size_t)M_TOTAL * K_TOTAL;
  __bf16* Wl = Wh + (size_t)N_TOTAL * K_TOTAL;

  yat_convx_kernel<<<M_TOTAL, 256, 0, stream>>>(x, Xh, Xl, xsq);
  yat_convw_kernel<<<dim3(K_TOTAL / 32, N_TOTAL / 32), 256, 0, stream>>>(w, Wh, Wl);
  yat_wsq_kernel<<<N_TOTAL / 256, 256, 0, stream>>>(w, wsq);

  dim3 grid(N_TOTAL / BN, M_TOTAL / BM);
  yat_gemm_kernel<<<grid, 256, 0, stream>>>(Xh, Xl, Wh, Wl, bias, alpha, xsq, wsq, out);
}
